// RNN_7464653160860
// MI455X (gfx1250) — compile-verified, matrix operand order checked by reading, not run on hardware
//
#include <hip/hip_runtime.h>
#include <hip/hip_bf16.h>
#include <math.h>

// ---------------------------------------------------------------------------
// MI455X (gfx1250) single-step 2-layer LSTM + 32000-way log-softmax.
// Bandwidth-bound mat-vec (~660MB fp32 weights / call, AI ~0.5 flop/B):
// stream weights once with non-temporal b128 loads, push the math through
// v_wmma_f32_16x16x32_bf16 using a hi/lo bf16 split for near-fp32 accuracy.
// ---------------------------------------------------------------------------

typedef __attribute__((ext_vector_type(16))) __bf16 v16bf;
typedef __attribute__((ext_vector_type(8)))  float  v8f;
typedef __attribute__((ext_vector_type(4)))  float  vf4;

#define MV_WAVES   8
#define MV_THREADS 256
#define MAX_K      4096

// Split fp32 into bf16 hi (round-to-nearest) + bf16 lo (residual).
__device__ __forceinline__ void split_bf16(float f, __bf16& hi, __bf16& lo) {
    unsigned u  = __float_as_uint(f);
    unsigned hu = (u + 0x8000u) & 0xFFFF0000u;       // RN to bf16 grid
    hi = __builtin_bit_cast(__bf16, (unsigned short)(hu >> 16));
    float r = f - __uint_as_float(hu);               // exact residual
    lo = __builtin_bit_cast(__bf16,
         (unsigned short)((__float_as_uint(r) + 0x8000u) >> 16));
}

// out[0..M) = W1 @ lds_vec[0..K1) + W2 @ lds_vec[K1..K1+K2) + b1 + b2
// lds_vec is the concatenation of up to three staged fp32 source vectors.
// One wave per 16-row output tile; K consumed 32/step via 3 bf16 WMMAs.
__global__ void __launch_bounds__(MV_THREADS)
wmma_matvec(const float* __restrict__ s0, int n0,
            const float* __restrict__ s1, int n1,
            const float* __restrict__ s2, int n2,
            const float* __restrict__ W1, int K1,
            const float* __restrict__ W2, int K2,
            const float* __restrict__ b1,
            const float* __restrict__ b2,
            float* __restrict__ out, int M)
{
    __shared__ __bf16 xhi[MAX_K];
    __shared__ __bf16 xlo[MAX_K];

    const int tid = threadIdx.x;
    for (int i = tid; i < n0; i += MV_THREADS) split_bf16(s0[i], xhi[i], xlo[i]);
    for (int i = tid; i < n1; i += MV_THREADS) split_bf16(s1[i], xhi[n0+i], xlo[n0+i]);
    if (s2)
        for (int i = tid; i < n2; i += MV_THREADS)
            split_bf16(s2[i], xhi[n0+n1+i], xlo[n0+n1+i]);
    __syncthreads();

    const int lane = tid & 31;            // wave32 on gfx1250
    const int wave = tid >> 5;
    const int half = lane >> 4;           // ISA A/B layout half-lane select
    const int mrow = lane & 15;           // row within 16-row tile
    const int tile = blockIdx.x * MV_WAVES + wave;
    if (tile * 16 >= M) return;

    v8f acc{};                            // 16x16 f32 C/D, 8 VGPRs

    #pragma unroll 1
    for (int seg = 0; seg < 2; ++seg) {
        const float* W = seg ? W2 : W1;
        const int    K = seg ? K2 : K1;
        const int xoff = seg ? K1 : 0;
        if (!W || K <= 0) continue;

        const float* rowp = W + (size_t)(tile * 16 + mrow) * (size_t)K;

        for (int kb = 0; kb < K; kb += 32) {
            // A tile rows: lane m -> K[o..o+7],K[o+16..o+23] (o = half*8).
            // Together lanes m and m+16 consume 128 contiguous bytes/row.
            const int o1 = kb + half * 8;
            vf4 aw[4];
            aw[0] = __builtin_nontemporal_load((const vf4*)(rowp + o1));
            aw[1] = __builtin_nontemporal_load((const vf4*)(rowp + o1 + 4));
            aw[2] = __builtin_nontemporal_load((const vf4*)(rowp + o1 + 16));
            aw[3] = __builtin_nontemporal_load((const vf4*)(rowp + o1 + 20));
            const float* wf = (const float*)aw;

            v16bf Ahi, Alo;
            #pragma unroll
            for (int e = 0; e < 16; ++e) {
                __bf16 h, l;
                split_bf16(wf[e], h, l);
                Ahi[e] = h; Alo[e] = l;
            }

            // B: all 16 columns identical = vector chunk (batch-1 broadcast).
            const int bo = xoff + o1;
            v16bf Bhi, Blo;
            #pragma unroll
            for (int e = 0; e < 8; ++e) {
                Bhi[e]     = xhi[bo + e];
                Bhi[e + 8] = xhi[bo + 16 + e];
                Blo[e]     = xlo[bo + e];
                Blo[e + 8] = xlo[bo + 16 + e];
            }

            // w·x ~= hi*hi + lo*hi + hi*lo  (fp32 accumulate in WMMA)
            acc = __builtin_amdgcn_wmma_f32_16x16x32_bf16(
                      false, Ahi, false, Bhi, (short)0, acc, false, false);
            acc = __builtin_amdgcn_wmma_f32_16x16x32_bf16(
                      false, Alo, false, Bhi, (short)0, acc, false, false);
            acc = __builtin_amdgcn_wmma_f32_16x16x32_bf16(
                      false, Ahi, false, Blo, (short)0, acc, false, false);
        }
    }

    // D layout: VGPR j, lane l -> M = j + 8*(l>=16), N = l%16. Take N==0.
    if (mrow == 0) {
        const int r0 = tile * 16 + half * 8;
        #pragma unroll
        for (int j = 0; j < 8; ++j) {
            const int r = r0 + j;
            float v = acc[j];
            if (b1) v += b1[r];
            if (b2) v += b2[r];
            out[r] = v;
        }
    }
}

// PyTorch gate order i,f,g,o on a 4H gate vector.
__global__ void __launch_bounds__(256)
lstm_cell(const float* __restrict__ gates, const float* __restrict__ cprev,
          float* __restrict__ h, float* __restrict__ c, int H)
{
    const int j = blockIdx.x * blockDim.x + threadIdx.x;
    if (j >= H) return;
    const float ig = gates[j];
    const float fg = gates[H + j];
    const float gg = gates[2 * H + j];
    const float og = gates[3 * H + j];
    const float i = 1.0f / (1.0f + expf(-ig));
    const float f = 1.0f / (1.0f + expf(-fg));
    const float g = tanhf(gg);
    const float o = 1.0f / (1.0f + expf(-og));
    const float cn = f * cprev[j] + i * g;
    c[j] = cn;
    h[j] = o * tanhf(cn);
}

// In-place log_softmax over n elements; single workgroup (n=32000 -> L2 hot).
__global__ void __launch_bounds__(1024)
log_softmax_inplace(float* __restrict__ x, int n)
{
    __shared__ float red[32];
    const int tid = threadIdx.x, lane = tid & 31, wid = tid >> 5;
    const int nw = blockDim.x >> 5;

    float m = -3.4e38f;
    for (int i = tid; i < n; i += blockDim.x) m = fmaxf(m, x[i]);
    for (int o = 16; o > 0; o >>= 1) m = fmaxf(m, __shfl_xor(m, o, 32));
    if (lane == 0) red[wid] = m;
    __syncthreads();
    if (wid == 0) {
        float t = (lane < nw) ? red[lane] : -3.4e38f;
        for (int o = 16; o > 0; o >>= 1) t = fmaxf(t, __shfl_xor(t, o, 32));
        if (lane == 0) red[0] = t;
    }
    __syncthreads();
    m = red[0];
    __syncthreads();

    float s = 0.0f;
    for (int i = tid; i < n; i += blockDim.x) s += expf(x[i] - m);
    for (int o = 16; o > 0; o >>= 1) s += __shfl_xor(s, o, 32);
    if (lane == 0) red[wid] = s;
    __syncthreads();
    if (wid == 0) {
        float t = (lane < nw) ? red[lane] : 0.0f;
        for (int o = 16; o > 0; o >>= 1) t += __shfl_xor(t, o, 32);
        if (lane == 0) red[0] = t;
    }
    __syncthreads();
    const float lse = m + logf(red[0]);

    for (int i = tid; i < n; i += blockDim.x) x[i] -= lse;
}

extern "C" void kernel_launch(void* const* d_in, const int* in_sizes, int n_in,
                              void* d_out, int out_size, void* d_ws, size_t ws_size,
                              hipStream_t stream)
{
    (void)in_sizes; (void)n_in; (void)out_size; (void)ws_size;

    const float* category = (const float*)d_in[0];   // [1024]
    const float* input    = (const float*)d_in[1];   // [1024]
    const float* hidden   = (const float*)d_in[2];   // [2,1,2048]
    const float* cellst   = (const float*)d_in[3];   // [2,1,2048]
    const float* w_ih_l0  = (const float*)d_in[4];   // [8192,2048]
    const float* w_hh_l0  = (const float*)d_in[5];   // [8192,2048]
    const float* b_ih_l0  = (const float*)d_in[6];
    const float* b_hh_l0  = (const float*)d_in[7];
    const float* w_ih_l1  = (const float*)d_in[8];   // [8192,2048]
    const float* w_hh_l1  = (const float*)d_in[9];
    const float* b_ih_l1  = (const float*)d_in[10];
    const float* b_hh_l1  = (const float*)d_in[11];
    const float* w_out    = (const float*)d_in[12];  // [32000,3072]
    const float* b_out    = (const float*)d_in[13];

    const int H = 2048, G = 4 * H, OUT = 32000;

    float* out   = (float*)d_out;
    float* logp  = out;                 // [32000]
    float* h1    = out + OUT;           // new_hidden[0]
    float* h2    = out + OUT + H;       // new_hidden[1]
    float* c1    = out + OUT + 2 * H;   // new_cell[0]
    float* c2    = out + OUT + 3 * H;   // new_cell[1]
    float* gates = (float*)d_ws;        // [8192] scratch (reused per layer)

    // Layer 0: gates = W_ih·[cat|in] + W_hh·h0 + b_ih + b_hh
    wmma_matvec<<<G / (16 * MV_WAVES), MV_THREADS, 0, stream>>>(
        category, 1024, input, 1024, hidden, H,
        w_ih_l0, 2048, w_hh_l0, 2048, b_ih_l0, b_hh_l0, gates, G);
    lstm_cell<<<(H + 255) / 256, 256, 0, stream>>>(gates, cellst, h1, c1, H);

    // Layer 1: gates = W_ih·h1 + W_hh·h_prev1 + biases
    wmma_matvec<<<G / (16 * MV_WAVES), MV_THREADS, 0, stream>>>(
        h1, H, hidden + H, H, nullptr, 0,
        w_ih_l1, 2048, w_hh_l1, 2048, b_ih_l1, b_hh_l1, gates, G);
    lstm_cell<<<(H + 255) / 256, 256, 0, stream>>>(gates, cellst + H, h2, c2, H);

    // Output: logits = W_out·[cat|h2] + b_out   (393MB stream — the hot loop)
    wmma_matvec<<<OUT / (16 * MV_WAVES), MV_THREADS, 0, stream>>>(
        category, 1024, h2, H, nullptr, 0,
        w_out, 3072, nullptr, 0, b_out, nullptr, logp, OUT);

    log_softmax_inplace<<<1, 1024, 0, stream>>>(logp, OUT);
}
